// TransformerEncoder_6305011990581
// MI455X (gfx1250) — compile-verified, matrix operand order checked by reading, not run on hardware
//
#include <hip/hip_runtime.h>
#include <math.h>

// ---------------------------------------------------------------------------
// Problem constants
// ---------------------------------------------------------------------------
#define DMODEL   768
#define NHEAD    12
#define DH       64
#define NLAYER   6
#define BATCH    4
#define NTOK     1024
#define MTOK     (BATCH * NTOK)       // 4096
#define HID      (4 * DMODEL)         // 3072
#define QKVD     (3 * DMODEL)         // 2304
#define LN_EPS   1e-5f

typedef __attribute__((ext_vector_type(16))) __bf16 v16bf;
typedef __attribute__((ext_vector_type(8)))  float  v8f;

union FragU { uint4 u[2]; v16bf f; };

// ---------------------------------------------------------------------------
// WMMA + async helpers
// ---------------------------------------------------------------------------
static __device__ __forceinline__ v8f wmma_bf16(v16bf a, v16bf b, v8f c) {
  return __builtin_amdgcn_wmma_f32_16x16x32_bf16(
      /*neg_a=*/false, a, /*neg_b=*/false, b,
      /*c_mod=*/(short)0, c, /*reuse_a=*/false, /*reuse_b=*/false);
}

// One wave-wide async global->LDS 16B-per-lane copy (CDNA5, ASYNCcnt).
static __device__ __forceinline__ void async_b128(const __bf16* g, __bf16* l) {
  const unsigned lofs = (unsigned)(size_t)l;  // generic addr low 32 = LDS offset
  asm volatile("global_load_async_to_lds_b128 %0, %1, off"
               :: "v"(lofs), "v"(g) : "memory");
}
static __device__ __forceinline__ void wait_async0() {
  asm volatile("s_wait_asynccnt 0x0" ::: "memory");
}

// Load a 16x32 bf16 fragment (A layout; B fragments come from transposed
// tiles so the same layout applies). ISA 16-bit A layout: lanes 0..15 ->
// K 0..7 & 16..23, lanes 16..31 -> K 8..15 & 24..31 => two b128s per lane.
static __device__ __forceinline__ v16bf load_frag(const __bf16* base, int row,
                                                  int stride) {
  const int lane = threadIdx.x & 31;
  const int r    = row + (lane & 15);
  const int kb   = (lane >> 4) * 8;
  const __bf16* p = base + (size_t)r * stride + kb;
  FragU t;
  t.u[0] = *(const uint4*)(p);
  t.u[1] = *(const uint4*)(p + 16);
  return t.f;
}

// ---------------------------------------------------------------------------
// LayerNorm: fp32 [MTOK][DMODEL] -> bf16 [MTOK][DMODEL]
// ---------------------------------------------------------------------------
__global__ __launch_bounds__(256) void layernorm_kernel(
    const float* __restrict__ x, const float* __restrict__ g,
    const float* __restrict__ b, __bf16* __restrict__ out) {
  const int row = blockIdx.x;
  const int t   = threadIdx.x;
  const float* xr = x + (size_t)row * DMODEL;
  __shared__ float red[256];

  float s = 0.f;
  for (int i = t; i < DMODEL; i += 256) s += xr[i];
  red[t] = s;
  __syncthreads();
  for (int k = 128; k > 0; k >>= 1) {
    if (t < k) red[t] += red[t + k];
    __syncthreads();
  }
  const float mean = red[0] * (1.0f / DMODEL);
  __syncthreads();

  float v = 0.f;
  for (int i = t; i < DMODEL; i += 256) {
    float d = xr[i] - mean;
    v += d * d;
  }
  red[t] = v;
  __syncthreads();
  for (int k = 128; k > 0; k >>= 1) {
    if (t < k) red[t] += red[t + k];
    __syncthreads();
  }
  const float inv = rsqrtf(red[0] * (1.0f / DMODEL) + LN_EPS);

  __bf16* orow = out + (size_t)row * DMODEL;
  for (int i = t; i < DMODEL; i += 256)
    orow[i] = (__bf16)((xr[i] - mean) * inv * g[i] + b[i]);
}

// ---------------------------------------------------------------------------
// Transpose + convert: W fp32 [K][N] -> Wt bf16 [N][K] (32x32 LDS tiles)
// ---------------------------------------------------------------------------
__global__ __launch_bounds__(256) void transpose_cvt_kernel(
    const float* __restrict__ W, __bf16* __restrict__ Wt, int K, int N) {
  __shared__ float tile[32][33];
  const int t    = threadIdx.x;
  const int nblk = N >> 5;
  const int k0   = (blockIdx.x / nblk) * 32;
  const int n0   = (blockIdx.x % nblk) * 32;
  for (int i = 0; i < 4; ++i) {
    const int e = t + i * 256, kk = e >> 5, nn = e & 31;
    tile[kk][nn] = W[(size_t)(k0 + kk) * N + n0 + nn];
  }
  __syncthreads();
  for (int i = 0; i < 4; ++i) {
    const int e = t + i * 256, nn = e >> 5, kk = e & 31;
    Wt[(size_t)(n0 + nn) * K + k0 + kk] = (__bf16)tile[kk][nn];
  }
}

// ---------------------------------------------------------------------------
// GEMM: C[MTOK][Nout] = A_bf16[MTOK][K] * Wt_bf16^T  (Wt stored [Nout][K])
//   mode 0: store bf16(acc + bias)
//   mode 1: store bf16(gelu(acc + bias))
//   mode 2: resid_f32 += acc + bias
// Workgroup tile 64x256, 8 waves of 32x64 (2x4 WMMA tiles). Double-buffered
// LDS filled by async global->LDS loads (ASYNCcnt) for pipelining.
// ---------------------------------------------------------------------------
__global__ __launch_bounds__(256) void gemm_bf16_kernel(
    const __bf16* __restrict__ A, const __bf16* __restrict__ Wt,
    const float* __restrict__ bias, __bf16* __restrict__ outb,
    float* __restrict__ resid, int K, int Nout, int mode) {
  __shared__ __bf16 At[2][64 * 32];     // [row][k]
  __shared__ __bf16 Ws[2][256 * 32];    // [n][k]

  const int t    = threadIdx.x;
  const int lane = t & 31;
  const int wave = t >> 5;
  const int wr   = wave >> 2;           // 0..1
  const int wc   = wave & 3;            // 0..3
  const int nblk = Nout >> 8;
  const int row0 = (blockIdx.x / nblk) * 64;
  const int col0 = (blockIdx.x % nblk) * 256;

  const int ar = t >> 2, aseg = t & 3;  // A: 64 rows x 4 b128 segs
  const __bf16* agp = A + (size_t)(row0 + ar) * K + aseg * 8;

  v8f acc[2][4];
  for (int i = 0; i < 2; ++i)
    for (int j = 0; j < 4; ++j) acc[i][j] = (v8f){};

  auto stage = [&](int k0, int buf) {
    async_b128(agp + k0, &At[buf][ar * 32 + aseg * 8]);
    for (int j = 0; j < 4; ++j) {
      const int e = t + j * 256, n = e >> 2, seg = e & 3;
      async_b128(Wt + (size_t)(col0 + n) * K + k0 + seg * 8,
                 &Ws[buf][n * 32 + seg * 8]);
    }
  };

  stage(0, 0);
  wait_async0();
  __syncthreads();

  const int ksteps = K >> 5;
  for (int i = 0; i < ksteps; ++i) {
    const int buf = i & 1;
    if (i + 1 < ksteps) stage((i + 1) << 5, buf ^ 1);  // overlap with compute

    v16bf a0 = load_frag(At[buf], wr * 32, 32);
    v16bf a1 = load_frag(At[buf], wr * 32 + 16, 32);
    for (int j = 0; j < 4; ++j) {
      v16bf bj = load_frag(Ws[buf], wc * 64 + j * 16, 32);
      acc[0][j] = wmma_bf16(a0, bj, acc[0][j]);
      acc[1][j] = wmma_bf16(a1, bj, acc[1][j]);
    }
    wait_async0();
    __syncthreads();
  }

  // Epilogue. C layout: lane L, vgpr v -> m = v + 8*(L/16), n = L%16.
  const int m_lo = (lane >> 4) * 8;
  const int nn   = lane & 15;
  for (int i = 0; i < 2; ++i)
    for (int j = 0; j < 4; ++j)
      for (int v = 0; v < 8; ++v) {
        const int gr = row0 + wr * 32 + i * 16 + m_lo + v;
        const int gc = col0 + wc * 64 + j * 16 + nn;
        float val = acc[i][j][v];
        if (bias) val += bias[gc];
        if (mode == 1)
          val = 0.5f * val * (1.0f + erff(val * 0.70710678118654752f));
        const size_t o = (size_t)gr * Nout + gc;
        if (mode == 2)
          resid[o] += val;
        else
          outb[o] = (__bf16)val;
      }
}

// ---------------------------------------------------------------------------
// Fused flash-style attention with spatial bias.
// qkv bf16 [MTOK][3*DMODEL]; per head slice of 64. Grid: B*H*(N/128) blocks,
// 8 waves per block, each wave owns 16 query rows.
// ---------------------------------------------------------------------------
__global__ __launch_bounds__(256) void attention_kernel(
    const __bf16* __restrict__ qkv, const float* __restrict__ sw,
    __bf16* __restrict__ attn_out) {
  const int idx = blockIdx.x;
  const int qb  = idx & 7;               // N/128 = 8
  const int h   = (idx >> 3) % NHEAD;
  const int b   = idx / (8 * NHEAD);
  const int t    = threadIdx.x;
  const int lane = t & 31;
  const int wave = t >> 5;
  const int q0   = qb * 128 + wave * 16; // query rows [q0, q0+16)
  const size_t tok0 = (size_t)b * NTOK;

  __shared__ __bf16 Ktile[32 * 64];      // [key][d]
  __shared__ __bf16 Vt[64 * 32];         // [d][key] (transposed)
  __shared__ __bf16 Pbuf[8 * 16 * 32];   // per-wave P tile

  // Preload Q fragments for this wave (two K-steps over DH=64).
  v16bf aq[2];
  for (int ks = 0; ks < 2; ++ks)
    aq[ks] = load_frag(qkv + tok0 * QKVD + h * DH + ks * 32, q0, QKVD);

  v8f o[4];
  for (int d = 0; d < 4; ++d) o[d] = (v8f){};
  float rmax[8], rsum[8];
  for (int v = 0; v < 8; ++v) { rmax[v] = -1e30f; rsum[v] = 0.f; }

  const float scale = 0.125f;            // 1/sqrt(64)
  const int   m_lo  = (lane >> 4) * 8;
  const int   ncol  = lane & 15;

  for (int kb0 = 0; kb0 < NTOK; kb0 += 32) {
    // Stage K block [32][64] (one b128 per thread).
    {
      const int kr = t >> 3, seg = t & 7;
      const uint4* src = (const uint4*)(qkv + (tok0 + kb0 + kr) * QKVD +
                                        DMODEL + h * DH + seg * 8);
      *(uint4*)(Ktile + kr * 64 + seg * 8) = *src;
    }
    // Stage V block transposed [64][32].
    for (int e = t; e < 2048; e += 256) {
      const int kr = e >> 6, d = e & 63;
      Vt[d * 32 + kr] =
          qkv[(tok0 + kb0 + kr) * QKVD + 2 * DMODEL + h * DH + d];
    }
    __syncthreads();

    // S = Q * K^T (two 16-key column tiles).
    v8f s[2];
    for (int ns = 0; ns < 2; ++ns) {
      v8f sa = (v8f){};
      for (int ks = 0; ks < 2; ++ks)
        sa = wmma_bf16(aq[ks], load_frag(Ktile + ks * 32, ns * 16, 64), sa);
      s[ns] = sa;
    }

    // Scale + spatial bias + online softmax (rows grouped per 16-lane half).
    float p0v[8], p1v[8];
    for (int v = 0; v < 8; ++v) {
      const int qrow = q0 + m_lo + v;
      const size_t swr = ((size_t)b * NTOK + qrow) * NTOK + kb0;
      float s0 = s[0][v] * scale + sw[swr + ncol];
      float s1 = s[1][v] * scale + sw[swr + 16 + ncol];
      float mx = fmaxf(s0, s1);
      for (int off = 1; off < 16; off <<= 1)
        mx = fmaxf(mx, __shfl_xor(mx, off, 32));
      const float nm    = fmaxf(rmax[v], mx);
      const float alpha = __expf(rmax[v] - nm);
      const float p0 = __expf(s0 - nm);
      const float p1 = __expf(s1 - nm);
      float ss = p0 + p1;
      for (int off = 1; off < 16; off <<= 1) ss += __shfl_xor(ss, off, 32);
      rsum[v] = rsum[v] * alpha + ss;
      rmax[v] = nm;
      for (int d = 0; d < 4; ++d) o[d][v] *= alpha;
      p0v[v] = p0;
      p1v[v] = p1;
    }

    // Re-layout P via wave-private LDS into an A fragment.
    __bf16* Pw = Pbuf + wave * (16 * 32);
    for (int v = 0; v < 8; ++v) {
      Pw[(m_lo + v) * 32 + ncol]      = (__bf16)p0v[v];
      Pw[(m_lo + v) * 32 + 16 + ncol] = (__bf16)p1v[v];
    }
    __syncthreads();
    const v16bf pf = load_frag(Pw, 0, 32);

    // O += P * V
    for (int d = 0; d < 4; ++d)
      o[d] = wmma_bf16(pf, load_frag(Vt, d * 16, 32), o[d]);
    __syncthreads();
  }

  // Normalize and store concatenated head outputs (bf16).
  for (int d = 0; d < 4; ++d)
    for (int v = 0; v < 8; ++v) {
      const int qrow = q0 + m_lo + v;
      attn_out[(tok0 + qrow) * DMODEL + h * DH + d * 16 + ncol] =
          (__bf16)(o[d][v] / rsum[v]);
    }
}

// ---------------------------------------------------------------------------
// Host launcher
// ---------------------------------------------------------------------------
extern "C" void kernel_launch(void* const* d_in, const int* in_sizes, int n_in,
                              void* d_out, int out_size, void* d_ws,
                              size_t ws_size, hipStream_t stream) {
  const float* x     = (const float*)d_in[0];
  const float* sw    = (const float*)d_in[1];
  const float* ln1_g = (const float*)d_in[2];
  const float* ln1_b = (const float*)d_in[3];
  const float* w_qkv = (const float*)d_in[4];
  const float* w_out = (const float*)d_in[5];
  const float* b_out = (const float*)d_in[6];
  const float* ln2_g = (const float*)d_in[7];
  const float* ln2_b = (const float*)d_in[8];
  const float* w1    = (const float*)d_in[9];
  const float* b1    = (const float*)d_in[10];
  const float* w2    = (const float*)d_in[11];
  const float* b2    = (const float*)d_in[12];

  // Workspace layout
  char* ws = (char*)d_ws;
  size_t off = 0;
  auto take = [&](size_t bytes) {
    char* p = ws + off;
    off += (bytes + 255) & ~(size_t)255;
    return p;
  };
  float*  x_f32 = (float*)take((size_t)MTOK * DMODEL * 4);
  __bf16* xn    = (__bf16*)take((size_t)MTOK * DMODEL * 2);
  __bf16* qkv   = (__bf16*)take((size_t)MTOK * QKVD * 2);
  __bf16* attn  = (__bf16*)take((size_t)MTOK * DMODEL * 2);
  __bf16* hmlp  = (__bf16*)take((size_t)MTOK * HID * 2);
  __bf16* wt    = (__bf16*)take((size_t)DMODEL * HID * 2);  // reusable [N][K]

  hipMemcpyAsync(x_f32, x, (size_t)MTOK * DMODEL * 4,
                 hipMemcpyDeviceToDevice, stream);

  const int g_qkv = (MTOK / 64) * (QKVD / 256);   // 64*9
  const int g_d   = (MTOK / 64) * (DMODEL / 256); // 64*3
  const int g_h   = (MTOK / 64) * (HID / 256);    // 64*12
  const int g_att = BATCH * NHEAD * (NTOK / 128); // 384
  const int c_qkv = (DMODEL / 32) * (QKVD / 32);
  const int c_d   = (DMODEL / 32) * (DMODEL / 32);
  const int c_h   = (DMODEL / 32) * (HID / 32);

  for (int l = 0; l < NLAYER; ++l) {
    layernorm_kernel<<<MTOK, 256, 0, stream>>>(
        x_f32, ln1_g + (size_t)l * DMODEL, ln1_b + (size_t)l * DMODEL, xn);

    transpose_cvt_kernel<<<c_qkv, 256, 0, stream>>>(
        w_qkv + (size_t)l * DMODEL * QKVD, wt, DMODEL, QKVD);
    gemm_bf16_kernel<<<g_qkv, 256, 0, stream>>>(xn, wt, nullptr, qkv, nullptr,
                                                DMODEL, QKVD, 0);

    attention_kernel<<<g_att, 256, 0, stream>>>(qkv, sw, attn);

    transpose_cvt_kernel<<<c_d, 256, 0, stream>>>(
        w_out + (size_t)l * DMODEL * DMODEL, wt, DMODEL, DMODEL);
    gemm_bf16_kernel<<<g_d, 256, 0, stream>>>(
        attn, wt, b_out + (size_t)l * DMODEL, nullptr, x_f32, DMODEL, DMODEL,
        2);

    layernorm_kernel<<<MTOK, 256, 0, stream>>>(
        x_f32, ln2_g + (size_t)l * DMODEL, ln2_b + (size_t)l * DMODEL, xn);

    transpose_cvt_kernel<<<c_h, 256, 0, stream>>>(
        w1 + (size_t)l * DMODEL * HID, wt, DMODEL, HID);
    gemm_bf16_kernel<<<g_h, 256, 0, stream>>>(
        xn, wt, b1 + (size_t)l * HID, hmlp, nullptr, DMODEL, HID, 1);

    transpose_cvt_kernel<<<c_h, 256, 0, stream>>>(
        w2 + (size_t)l * HID * DMODEL, wt, HID, DMODEL);
    gemm_bf16_kernel<<<g_d, 256, 0, stream>>>(
        hmlp, wt, b2 + (size_t)l * DMODEL, nullptr, x_f32, HID, DMODEL, 2);
  }

  hipMemcpyAsync(d_out, x_f32, (size_t)MTOK * DMODEL * 4,
                 hipMemcpyDeviceToDevice, stream);
}
